// MultiHeadAttentionLayer_77567109366402
// MI455X (gfx1250) — matrix operand order repaired and checked by reading, not compiled, not run on hardware
//
#include <hip/hip_runtime.h>
#include <hip/hip_bf16.h>

typedef __attribute__((ext_vector_type(16))) _Float16 v16h;
typedef __attribute__((ext_vector_type(8)))  _Float16 v8h;
typedef __attribute__((ext_vector_type(4)))  _Float16 v4h;
typedef __attribute__((ext_vector_type(4)))  float    f4;
typedef __attribute__((ext_vector_type(8)))  float    v8f;

#define BB   2
#define SS   2048
#define HIDD 1024
#define NHH  16
#define HDD  64
#define NEGV (-1e10f)

// 256 threads = 8 wave32s: 4 waves in M (BM=128), 2 waves in N (BNP template).
#define BM  128
#define BKT 64
#define LDT (BKT + 8)   // LDS row stride in halves; 144B rows (16B aligned)

// ---------------------------------------------------------------------------
// WMMA fragment loaders per CDNA5 ISA 7.12.2 (wave32).
// ---------------------------------------------------------------------------
__device__ inline v16h load_a_frag(const _Float16* base, int lane) {
    const int m  = lane & 15;
    const int kg = (lane >> 4) << 3;
    union { v16h v; unsigned u[8]; } r;
    const unsigned* row = (const unsigned*)(base + m * LDT);
#pragma unroll
    for (int i = 0; i < 8; ++i) {
        const int k = ((i < 4) ? (2 * i) : (2 * i + 8)) + kg;
        r.u[i] = row[k >> 1];
    }
    return r.v;
}

__device__ inline v16h load_b_frag(const _Float16* base, int lane) {
    const int n  = lane & 15;
    const int kb = (lane >> 4) << 3;
    union { v16h v; unsigned u[8]; } r;
    const unsigned* row = (const unsigned*)(base + n * LDT);
#pragma unroll
    for (int i = 0; i < 8; ++i) r.u[i] = row[(((i < 4) ? (2 * i) : (2 * i + 8)) + kb) >> 1];
    return r.v;
}

__device__ inline v8f wmma_f16(v16h a, v16h b, v8f c) {
    return __builtin_amdgcn_wmma_f32_16x16x32_f16(
        false, a, false, b, (short)0, c, false, false);
}

// async global->LDS 16B copy (per lane), tracked by ASYNCcnt (ISA §15.18.3 #98)
__device__ inline void async_copy_b128(void* lds_dst, const void* gsrc) {
    const unsigned lds_off = (unsigned)(unsigned long long)lds_dst;
    asm volatile("global_load_async_to_lds_b128 %0, %1, off"
                 :: "v"(lds_off), "v"(gsrc) : "memory");
}
__device__ inline void wait_asynccnt0() {
    asm volatile("s_wait_asynccnt 0" ::: "memory");
}

// ---------------------------------------------------------------------------
// Batched GEMM: Out[z] = X[z] @ W[z]^T (+bias) (+key-mask). f16 WMMA, f32 acc.
//   X element (m,k): X[z*bsx + m*ldx + k]
//   W element (n,k): W[z*bsw + n*wrs + k*wks]   (KS1 => wks==1, vectorized)
// OMODE: 0 plain Out[z*bso + m*ldo + n]
//        1 scatter [B*S,HID] -> head-major [B,NH,S,HD]
//        2 scatter per-(b,h) [S,HD] -> [B,S,HID]
// ASYNCS (requires XH&&WH&&KS1): stage LDS tiles with async global->LDS b128.
// ---------------------------------------------------------------------------
template<int BNP, bool XH, bool WH, bool OH, int OMODE, bool KS1, bool HASB,
         bool HASM, bool ASYNCS>
__global__ __launch_bounds__(256)
void wmma_gemm(const void* __restrict__ Xv, long ldx, long bsx,
               const void* __restrict__ Wv, long wrs, long wks, long bsw,
               const float* __restrict__ bias,
               void* __restrict__ Outv, long ldo, long bso,
               const int* __restrict__ mask,
               int K)
{
    constexpr int NSUB = BNP / 32;      // 16-wide n-subtiles per wave
    __shared__ _Float16 As[BM][LDT];
    __shared__ _Float16 Bs[BNP][LDT];

    const int tid  = threadIdx.x;
    const int lane = tid & 31;
    const int wave = tid >> 5;
    const int wm   = wave >> 1;         // 0..3
    const int wn   = wave & 1;          // 0..1
    const long z   = blockIdx.z;
    const int m0   = blockIdx.y * BM;
    const int n0   = blockIdx.x * BNP;

    const float*    Xf = (const float*)Xv + z * bsx;
    const _Float16* Xh = (const _Float16*)Xv + z * bsx;
    const float*    Wf = (const float*)Wv + z * bsw;
    const _Float16* Wh = (const _Float16*)Wv + z * bsw;

    v8f acc[2][NSUB] = {};

    for (int kt = 0; kt < K; kt += BKT) {
        // ---- stage X tile (BM x BKT halves)
        if (XH) {
#pragma unroll
            for (int i = 0; i < (BM * BKT) / (256 * 8); ++i) {
                const int e = i * 256 + tid;
                const int r = e >> 3;
                const int c = (e & 7) * 8;
                const _Float16* src = Xh + (long)(m0 + r) * ldx + kt + c;
                if (ASYNCS) async_copy_b128(&As[r][c], src);
                else        *(v8h*)&As[r][c] = *(const v8h*)src;
            }
        } else {
#pragma unroll
            for (int i = 0; i < (BM * BKT) / (256 * 4); ++i) {
                const int e = i * 256 + tid;
                const int r = e >> 4;
                const int c = (e & 15) * 4;
                const f4 f = *(const f4*)(Xf + (long)(m0 + r) * ldx + kt + c);
                v4h h;
                h.x = (_Float16)f.x; h.y = (_Float16)f.y;
                h.z = (_Float16)f.z; h.w = (_Float16)f.w;
                *(v4h*)&As[r][c] = h;
            }
        }
        // ---- stage W tile (BNP x BKT halves)
        if (KS1) {
            if (WH) {
#pragma unroll
                for (int i = 0; i < (BNP * BKT) / (256 * 8); ++i) {
                    const int e = i * 256 + tid;
                    const int r = e >> 3;
                    const int c = (e & 7) * 8;
                    const _Float16* src = Wh + (long)(n0 + r) * wrs + kt + c;
                    if (ASYNCS) async_copy_b128(&Bs[r][c], src);
                    else        *(v8h*)&Bs[r][c] = *(const v8h*)src;
                }
            } else {
#pragma unroll
                for (int i = 0; i < (BNP * BKT) / (256 * 4); ++i) {
                    const int e = i * 256 + tid;
                    const int r = e >> 4;
                    const int c = (e & 15) * 4;
                    const f4 f = *(const f4*)(Wf + (long)(n0 + r) * wrs + kt + c);
                    v4h h;
                    h.x = (_Float16)f.x; h.y = (_Float16)f.y;
                    h.z = (_Float16)f.z; h.w = (_Float16)f.w;
                    *(v4h*)&Bs[r][c] = h;
                }
            }
        } else {
            // strided k (V in AV GEMM): n-contiguous => coalesced u16 loads
#pragma unroll
            for (int i = 0; i < (BNP * BKT) / 256; ++i) {
                const int e = i * 256 + tid;
                const int n = e % BNP;
                const int k = e / BNP;
                Bs[n][k] = Wh[(long)(n0 + n) * wrs + (long)(kt + k) * wks];
            }
        }
        if (ASYNCS) wait_asynccnt0();
        __syncthreads();

#pragma unroll
        for (int ks = 0; ks < BKT; ks += 32) {
            const v16h a0 = load_a_frag(&As[wm * 32 +  0][ks], lane);
            const v16h a1 = load_a_frag(&As[wm * 32 + 16][ks], lane);
#pragma unroll
            for (int sn = 0; sn < NSUB; ++sn) {
                const v16h b = load_b_frag(&Bs[wn * (BNP / 2) + sn * 16][ks], lane);
                acc[0][sn] = wmma_f16(a0, b, acc[0][sn]);
                acc[1][sn] = wmma_f16(a1, b, acc[1][sn]);
            }
        }
        __syncthreads();
    }

    // ---- epilogue: bias / mask / layout scatter / store
    float*    Of = (float*)Outv;
    _Float16* Oh = (_Float16*)Outv;
#pragma unroll
    for (int sm = 0; sm < 2; ++sm) {
#pragma unroll
        for (int sn = 0; sn < NSUB; ++sn) {
            const int nn = n0 + wn * (BNP / 2) + sn * 16 + (lane & 15);
            float bv = 0.0f;
            if (HASB) bv = bias[nn];
            bool kill = false;
            if (HASM) kill = (mask[(z / NHH) * SS + nn] == 0);
#pragma unroll
            for (int j = 0; j < 8; ++j) {
                const int mm = m0 + wm * 32 + sm * 16 + ((lane >> 4) << 3) + j;
                float v = acc[sm][sn][j] + bv;
                if (HASM && kill) v = NEGV;
                long idx;
                if (OMODE == 0) {
                    idx = z * bso + (long)mm * ldo + nn;
                } else if (OMODE == 1) {
                    const int b = mm >> 11, s = mm & (SS - 1);
                    const int h = nn >> 6,  d = nn & (HDD - 1);
                    idx = (((long)b * NHH + h) * SS + s) * HDD + d;
                } else {
                    const int b = (int)z >> 4, h = (int)z & (NHH - 1);
                    idx = ((long)b * SS + mm) * (long)HIDD + h * HDD + nn;
                }
                if (OH) Oh[idx] = (_Float16)v;
                else    Of[idx] = v;
            }
        }
    }
}

// ---------------------------------------------------------------------------
// Fold per-head multiplicative linear into Q projection weights.
// ---------------------------------------------------------------------------
__global__ __launch_bounds__(256)
void fold_wq(const float* __restrict__ Wq, const float* __restrict__ bq,
             const float* __restrict__ Ww, const float* __restrict__ bw,
             _Float16* __restrict__ Wqp, float* __restrict__ bqp)
{
    const int k = blockIdx.x * 256 + threadIdx.x;
    const int n = blockIdx.y;
    const int h = n >> 6, d = n & (HDD - 1);
    float s = 0.0f;
#pragma unroll 4
    for (int dp = 0; dp < HDD; ++dp)
        s += Ww[d * HDD + dp] * Wq[(long)(h * HDD + dp) * HIDD + k];
    Wqp[(long)n * HIDD + k] = (_Float16)s;
    if (blockIdx.x == 0 && threadIdx.x == 0) {
        float t = bw[d];
        for (int dp = 0; dp < HDD; ++dp)
            t += Ww[d * HDD + dp] * bq[h * HDD + dp];
        bqp[n] = t;
    }
}

// ---------------------------------------------------------------------------
// In-place row softmax over the attention output region.
// ---------------------------------------------------------------------------
__global__ __launch_bounds__(256)
void softmax_rows(float* __restrict__ att, int cols)
{
    __shared__ float red[256];
    const long row = blockIdx.x;
    float* p = att + row * (long)cols;

    float mx = -3.0e38f;
    for (int c = threadIdx.x; c < cols; c += 256) mx = fmaxf(mx, p[c]);
    red[threadIdx.x] = mx; __syncthreads();
    for (int s = 128; s > 0; s >>= 1) {
        if (threadIdx.x < s) red[threadIdx.x] = fmaxf(red[threadIdx.x], red[threadIdx.x + s]);
        __syncthreads();
    }
    mx = red[0]; __syncthreads();

    float sum = 0.0f;
    for (int c = threadIdx.x; c < cols; c += 256) {
        const float e = __expf(p[c] - mx);
        p[c] = e;
        sum += e;
    }
    red[threadIdx.x] = sum; __syncthreads();
    for (int s = 128; s > 0; s >>= 1) {
        if (threadIdx.x < s) red[threadIdx.x] += red[threadIdx.x + s];
        __syncthreads();
    }
    const float inv = 1.0f / red[0];
    for (int c = threadIdx.x; c < cols; c += 256) p[c] *= inv;
}

// ---------------------------------------------------------------------------
extern "C" void kernel_launch(void* const* d_in, const int* in_sizes, int n_in,
                              void* d_out, int out_size, void* d_ws, size_t ws_size,
                              hipStream_t stream)
{
    (void)in_sizes; (void)n_in; (void)out_size; (void)ws_size;

    const float* query = (const float*)d_in[0];
    const float* key_  = (const float*)d_in[1];
    const float* value = (const float*)d_in[2];
    const int*   mask  = (const int*)d_in[3];
    const float* Wq = (const float*)d_in[4];
    const float* bq = (const float*)d_in[5];
    const float* Wk = (const float*)d_in[6];
    const float* bk = (const float*)d_in[7];
    const float* Wv = (const float*)d_in[8];
    const float* bv = (const float*)d_in[9];
    const float* Ww = (const float*)d_in[10];
    const float* bw = (const float*)d_in[11];
    const float* Wo = (const float*)d_in[12];
    const float* bo = (const float*)d_in[13];

    float* x_out = (float*)d_out;                 // [B,S,HID]
    float* att   = x_out + (long)BB * SS * HIDD;  // [B,NH,S,S]

    char* ws = (char*)d_ws;
    _Float16* Wqp = (_Float16*)ws;  ws += (long)HIDD * HIDD * sizeof(_Float16);
    float*    bqp = (float*)ws;     ws += (long)HIDD * sizeof(float);
    _Float16* Qh  = (_Float16*)ws;  ws += (long)BB * SS * HIDD * sizeof(_Float16);
    _Float16* Kh  = (_Float16*)ws;  ws += (long)BB * SS * HIDD * sizeof(_Float16);
    _Float16* Vh  = (_Float16*)ws;  ws += (long)BB * SS * HIDD * sizeof(_Float16);
    float*    ctx = (float*)ws;     ws += (long)BB * SS * HIDD * sizeof(float);

    const dim3 blk(256);

    // 0) fold Ww/bw into the Q projection weights
    fold_wq<<<dim3(HIDD / 256, HIDD), blk, 0, stream>>>(Wq, bq, Ww, bw, Wqp, bqp);

    // 1) projections -> head-major f16 [B,NH,S,HD]   (BNP=128 tiles)
    const dim3 gproj(HIDD / 128, (BB * SS) / BM, 1);
    wmma_gemm<128, false, true,  true, 1, true, true, false, false>
        <<<gproj, blk, 0, stream>>>(query, HIDD, 0, Wqp, HIDD, 1, 0, bqp,
                                    Qh, 0, 0, nullptr, HIDD);
    wmma_gemm<128, false, false, true, 1, true, true, false, false>
        <<<gproj, blk, 0, stream>>>(key_,  HIDD, 0, Wk,  HIDD, 1, 0, bk,
                                    Kh, 0, 0, nullptr, HIDD);
    wmma_gemm<128, false, false, true, 1, true, true, false, false>
        <<<gproj, blk, 0, stream>>>(value, HIDD, 0, Wv,  HIDD, 1, 0, bv,
                                    Vh, 0, 0, nullptr, HIDD);

    // 2) energy = Qw @ K^T with key mask -> d_out attention slab (pre-softmax)
    //    f16/f16 inputs: async global->LDS staging path
    const dim3 ge(SS / 128, SS / BM, BB * NHH);
    wmma_gemm<128, true, true, false, 0, true, false, true, true>
        <<<ge, blk, 0, stream>>>(Qh, HDD, (long)SS * HDD, Kh, HDD, 1,
                                 (long)SS * HDD, nullptr,
                                 att, SS, (long)SS * SS, mask, HDD);

    // 3) softmax in place over the final attention output
    softmax_rows<<<BB * NHH * SS, blk, 0, stream>>>(att, SS);

    // 4) ctx = attention @ V, un-scattered to [B,S,HID]   (BNP=64, N=HD)
    const dim3 gav(1, SS / BM, BB * NHH);
    wmma_gemm<64, false, true, false, 2, false, false, false, false>
        <<<gav, blk, 0, stream>>>(att, SS, (long)SS * SS, Vh, 1, HDD,
                                  (long)SS * HDD, nullptr,
                                  ctx, 0, 0, nullptr, SS);

    // 5) x = ctx @ Wo^T + bo
    wmma_gemm<128, false, false, false, 0, true, true, false, false>
        <<<gproj, blk, 0, stream>>>(ctx, HIDD, 0, Wo, HIDD, 1, 0, bo,
                                    x_out, HIDD, 0, nullptr, HIDD);
}
